// EquivariantHON_88347477279279
// MI455X (gfx1250) — compile-verified, matrix operand order checked by reading, not run on hardware
//
#include <hip/hip_runtime.h>
#include <math.h>

// ---------------------------------------------------------------------------
// Problem constants (from reference)
// ---------------------------------------------------------------------------
#define CN1 100000
#define CN2 200000
#define CN3 100000
#define CM1 400000   // 2*N2
#define CM2 300000   // 3*N3
#define CB  64
#define CD_IN 32
#define CD_H  128
#define CMSG  64
#define COUT  10

typedef __attribute__((ext_vector_type(16))) _Float16 v16h;
typedef __attribute__((ext_vector_type(8)))  float    v8f;

// ---------------------------------------------------------------------------
// WMMA fragment helpers (wave32, CDNA5 layouts per CDNA5 ISA)
// A 16x32 f16: lane L holds row (L&15); elements 0..7  = K base..base+7,
//              elements 8..15 = K base+16..base+23, base = (L<16)?0:8.
// ---------------------------------------------------------------------------
static __device__ __forceinline__ v16h load_a_frag_f32(const float* __restrict__ rowp,
                                                       int kbase, int lane) {
  int kb = kbase + ((lane & 16) ? 8 : 0);
  v16h a;
#pragma unroll
  for (int i = 0; i < 8; ++i) a[i] = (_Float16)rowp[kb + i];
#pragma unroll
  for (int i = 0; i < 8; ++i) a[8 + i] = (_Float16)rowp[kb + 16 + i];
  return a;
}

// B fragment from global packed layout (32 f16 per (kt,n) group)
static __device__ __forceinline__ v16h load_b_frag(const _Float16* __restrict__ packed,
                                                   int kt, int Np, int ncol, int lane) {
  const _Float16* p = packed + ((size_t)(kt * Np + ncol) * 32) + ((lane & 16) ? 16 : 0);
  v16h b;
#pragma unroll
  for (int i = 0; i < 16; ++i) b[i] = p[i];
  return b;
}

// B fragment from LDS-staged packed layout, 17-dword padded stride per group
// (gcd(17,64)=1 -> banks well spread for the two b128 reads per lane)
static __device__ __forceinline__ v16h load_b_frag_lds(const uint32_t* ldsB,
                                                       int kt, int Np, int ncol, int lane) {
  const _Float16* p = (const _Float16*)(ldsB + (kt * Np + ncol) * 17 + ((lane & 16) ? 8 : 0));
  v16h b;
#pragma unroll
  for (int i = 0; i < 16; ++i) b[i] = p[i];
  return b;
}

static __device__ __forceinline__ float silu_f(float v) {
  return v / (1.0f + __expf(-v));
}

// ---------------------------------------------------------------------------
// Pack fp32 W[K x N] (row-major) into f16 B-fragment-major layout:
// out[((kt*Np + n)*32) + pos], pos 0..15 -> lane-half 0 K order {0..7,16..23},
// pos 16..31 -> lane-half 1 K order {8..15,24..31}. Zero pad k>=K, n>=N.
// ---------------------------------------------------------------------------
__global__ void pack_b_kernel(const float* __restrict__ W, _Float16* __restrict__ out,
                              int K, int N, int Np, int KT) {
  int idx = blockIdx.x * blockDim.x + threadIdx.x;
  int total = KT * Np * 32;
  if (idx >= total) return;
  int pos = idx & 31;
  int n = (idx >> 5) % Np;
  int kt = idx / (32 * Np);
  int kk;
  if (pos < 8)       kk = pos;           // 0..7
  else if (pos < 16) kk = pos + 8;       // 16..23
  else if (pos < 24) kk = pos - 8;       // 8..15
  else               kk = pos;           // 24..31
  int k = kt * 32 + kk;
  float v = (k < K && n < N) ? W[(size_t)k * N + n] : 0.0f;
  out[idx] = (_Float16)v;
}

// ---------------------------------------------------------------------------
// K-tile loop helper: one A fragment feeds NT accumulators (A loaded once).
// ---------------------------------------------------------------------------
template <int NT, bool LDSB>
static __device__ __forceinline__ void kloop(v8f (&c)[NT], const float* __restrict__ row,
                                             int nk, int& kt, const uint32_t* ldsB,
                                             const _Float16* __restrict__ Wp,
                                             int r, int lane) {
  for (int t = 0; t < nk; ++t, ++kt) {
    v16h a = load_a_frag_f32(row, t * 32, lane);
#pragma unroll
    for (int nt = 0; nt < NT; ++nt) {
      v16h b = LDSB ? load_b_frag_lds(ldsB, kt, NT * 16, nt * 16 + r, lane)
                    : load_b_frag(Wp, kt, NT * 16, nt * 16 + r, lane);
      c[nt] = __builtin_amdgcn_wmma_f32_16x16x32_f16(false, a, false, b, (short)0,
                                                     c[nt], false, false);
    }
  }
}

// ---------------------------------------------------------------------------
// WMMA GEMM, wave-per-M-tile: one wave computes a full 16 x (NT*16) strip so
// each A fragment is gathered exactly once. Up to 3 concatenated fp32 sources
// (widths multiples of 32). Weights staged in LDS when KT*Np <= 512 groups.
// ---------------------------------------------------------------------------
template <int NT>
__global__ __launch_bounds__(256)
void gemm_cat3_nt_kernel(const float* __restrict__ S0, int w0,
                         const float* __restrict__ S1, int w1,
                         const float* __restrict__ S2, int w2,
                         const _Float16* __restrict__ Wp, const float* __restrict__ bias,
                         float* __restrict__ Y, int M, int N, int K, int act) {
  __shared__ uint32_t ldsB[512 * 17];  // 34816 B
  const int Np = NT * 16;
  int lane = threadIdx.x & 31;
  int wv = threadIdx.x >> 5;
  int KT = K >> 5;

  bool useLds = (KT * Np) <= 512;
  if (useLds) {
    int nd = KT * Np * 16;
    const uint32_t* src = (const uint32_t*)Wp;
    for (int idx = threadIdx.x; idx < nd; idx += 256)
      ldsB[(idx >> 4) * 17 + (idx & 15)] = src[idx];
  }
  __syncthreads();

  int tm = (M + 15) >> 4;
  int tmi = blockIdx.x * 8 + wv;
  if (tmi >= tm) return;
  int r = lane & 15, half = lane >> 4;
  int mr = tmi * 16 + r;
  if (mr >= M) mr = M - 1;

  v8f c[NT];
#pragma unroll
  for (int nt = 0; nt < NT; ++nt) c[nt] = (v8f){};

  int kt = 0;
  if (useLds) {
    kloop<NT, true>(c, S0 + (size_t)mr * w0, w0 >> 5, kt, ldsB, Wp, r, lane);
    if (w1) kloop<NT, true>(c, S1 + (size_t)mr * w1, w1 >> 5, kt, ldsB, Wp, r, lane);
    if (w2) kloop<NT, true>(c, S2 + (size_t)mr * w2, w2 >> 5, kt, ldsB, Wp, r, lane);
  } else {
    kloop<NT, false>(c, S0 + (size_t)mr * w0, w0 >> 5, kt, ldsB, Wp, r, lane);
    if (w1) kloop<NT, false>(c, S1 + (size_t)mr * w1, w1 >> 5, kt, ldsB, Wp, r, lane);
    if (w2) kloop<NT, false>(c, S2 + (size_t)mr * w2, w2 >> 5, kt, ldsB, Wp, r, lane);
  }

#pragma unroll
  for (int nt = 0; nt < NT; ++nt) {
    int ncol = nt * 16 + r;
    float bb = (ncol < N) ? bias[ncol] : 0.0f;
#pragma unroll
    for (int j = 0; j < 8; ++j) {
      int m = tmi * 16 + j + half * 8;
      float v = c[nt][j] + bb;
      if (act == 1)      v = silu_f(v);
      else if (act == 2) v = fmaxf(v, 0.0f);
      if (m < M && ncol < N) Y[(size_t)m * N + ncol] = v;
    }
  }
}

// ---------------------------------------------------------------------------
// Generic wave-per-(Mtile,Ntile) GEMM for shapes the templated kernel doesn't
// cover (head layer with N=256). M is tiny there, so A re-gather is harmless.
// ---------------------------------------------------------------------------
__global__ __launch_bounds__(256)
void gemm_cat3_gen_kernel(const float* __restrict__ S0, int w0,
                          const float* __restrict__ S1, int w1,
                          const float* __restrict__ S2, int w2,
                          const _Float16* __restrict__ Wp, const float* __restrict__ bias,
                          float* __restrict__ Y, int M, int N, int Np, int K, int act) {
  int lane = threadIdx.x & 31;
  int wv = threadIdx.x >> 5;
  int tn = Np >> 4;
  int tm = (M + 15) >> 4;
  int wid = blockIdx.x * 8 + wv;
  if (wid >= tm * tn) return;
  int tmi = wid / tn, tni = wid % tn;
  int r = lane & 15, half = lane >> 4;
  int mr = tmi * 16 + r;
  if (mr >= M) mr = M - 1;
  int ncol = tni * 16 + r;

  v8f c = {};
  int kt = 0;
  {
    const float* row0 = S0 + (size_t)mr * w0;
    for (int t = 0; t < (w0 >> 5); ++t, ++kt) {
      v16h a = load_a_frag_f32(row0, t * 32, lane);
      v16h b = load_b_frag(Wp, kt, Np, ncol, lane);
      c = __builtin_amdgcn_wmma_f32_16x16x32_f16(false, a, false, b, (short)0, c, false, false);
    }
  }
  if (w1) {
    const float* row1 = S1 + (size_t)mr * w1;
    for (int t = 0; t < (w1 >> 5); ++t, ++kt) {
      v16h a = load_a_frag_f32(row1, t * 32, lane);
      v16h b = load_b_frag(Wp, kt, Np, ncol, lane);
      c = __builtin_amdgcn_wmma_f32_16x16x32_f16(false, a, false, b, (short)0, c, false, false);
    }
  }
  if (w2) {
    const float* row2 = S2 + (size_t)mr * w2;
    for (int t = 0; t < (w2 >> 5); ++t, ++kt) {
      v16h a = load_a_frag_f32(row2, t * 32, lane);
      v16h b = load_b_frag(Wp, kt, Np, ncol, lane);
      c = __builtin_amdgcn_wmma_f32_16x16x32_f16(false, a, false, b, (short)0, c, false, false);
    }
  }

  float bb = (ncol < N) ? bias[ncol] : 0.0f;
#pragma unroll
  for (int j = 0; j < 8; ++j) {
    int m = tmi * 16 + j + half * 8;
    float v = c[j] + bb;
    if (act == 1)      v = silu_f(v);
    else if (act == 2) v = fmaxf(v, 0.0f);
    if (m < M && ncol < N) Y[(size_t)m * N + ncol] = v;
  }
}

// ---------------------------------------------------------------------------
// Fused message kernel. One wave handles a tile of 16 incidences:
//   feat = [h_recv[recv](128) | h_nbr[nbr](128) | d2]  (K=256 WMMA + rank-1 d2)
//   hidden = silu(feat @ W1 + b1)   -> wave-private LDS (f16)
//   m      = silu(hidden @ W2 + b2) -> atomic scatter into msg_sum
//   c      = m @ wc + bc            -> dx_sum += diff*c ; cnt += 1
// W1/W2 fragments staged once per block in LDS.
// ---------------------------------------------------------------------------
__global__ __launch_bounds__(256)
void message_kernel(const float* __restrict__ h_recv, const float* __restrict__ h_nbr,
                    const float* __restrict__ x_recv, const float* __restrict__ x_nbr,
                    const int* __restrict__ recv_idx, const int* __restrict__ nbr_idx,
                    int Minc,
                    const _Float16* __restrict__ w1p, const float* __restrict__ w1_last,
                    const float* __restrict__ b1,
                    const _Float16* __restrict__ w2p, const float* __restrict__ b2,
                    const float* __restrict__ wc, const float* __restrict__ bc,
                    float* __restrict__ msg_sum, float* __restrict__ dx_sum,
                    float* __restrict__ cnt) {
  __shared__ uint32_t ldsW1[512 * 17];            // 8 kt * 64 n groups, 34816 B
  __shared__ uint32_t ldsW2[128 * 17];            // 2 kt * 64 n groups,  8704 B
  __shared__ _Float16 hl16[8][16][64];            // per-wave hidden, 16384 B

  // ---- cooperative weight staging ----
  {
    const uint32_t* s1 = (const uint32_t*)w1p;
    for (int idx = threadIdx.x; idx < 512 * 16; idx += 256)
      ldsW1[(idx >> 4) * 17 + (idx & 15)] = s1[idx];
    const uint32_t* s2 = (const uint32_t*)w2p;
    for (int idx = threadIdx.x; idx < 128 * 16; idx += 256)
      ldsW2[(idx >> 4) * 17 + (idx & 15)] = s2[idx];
  }
  __syncthreads();

  int lane = threadIdx.x & 31;
  int wv = threadIdx.x >> 5;
  int tiles = (Minc + 15) >> 4;
  int tile = blockIdx.x * 8 + wv;
  if (tile >= tiles) return;
  int base = tile << 4;
  int r = lane & 15;
  int half = lane >> 4;
  int i = base + r;
  bool valid = i < Minc;
  int rec = valid ? recv_idx[i] : 0;
  int nb  = valid ? nbr_idx[i] : 0;
  __builtin_prefetch(recv_idx + base + 128, 0, 1);
  __builtin_prefetch(nbr_idx + base + 128, 0, 1);

  float dxv = x_recv[rec * 3 + 0] - x_nbr[nb * 3 + 0];
  float dyv = x_recv[rec * 3 + 1] - x_nbr[nb * 3 + 1];
  float dzv = x_recv[rec * 3 + 2] - x_nbr[nb * 3 + 2];
  float d2 = dxv * dxv + dyv * dyv + dzv * dzv;

  const float* hr = h_recv + (size_t)rec * CD_H;
  const float* hn = h_nbr + (size_t)nb * CD_H;
  v16h a1[8];
#pragma unroll
  for (int kt = 0; kt < 4; ++kt) a1[kt] = load_a_frag_f32(hr, kt * 32, lane);
#pragma unroll
  for (int kt = 0; kt < 4; ++kt) a1[4 + kt] = load_a_frag_f32(hn, kt * 32, lane);

  // GEMM1: [16x256] @ [256x64] + d2 rank-1 + bias, silu -> LDS (f16)
#pragma unroll
  for (int nt = 0; nt < 4; ++nt) {
    v8f c = {};
#pragma unroll
    for (int kt = 0; kt < 8; ++kt) {
      v16h b = load_b_frag_lds(ldsW1, kt, 64, nt * 16 + r, lane);
      c = __builtin_amdgcn_wmma_f32_16x16x32_f16(false, a1[kt], false, b, (short)0,
                                                 c, false, false);
    }
    int n = nt * 16 + r;
    float wl = w1_last[n];
    float bb = b1[n];
#pragma unroll
    for (int j = 0; j < 8; ++j) {
      int m = j + half * 8;
      float d2m = __shfl(d2, m, 32);
      float v = c[j] + d2m * wl + bb;
      hl16[wv][m][n] = (_Float16)silu_f(v);
    }
  }
  asm volatile("s_wait_dscnt 0x0" ::: "memory");  // order wave-private LDS RAW

  // A fragments for GEMM2 from LDS hidden (already f16)
  v16h a2[2];
#pragma unroll
  for (int kt = 0; kt < 2; ++kt) {
    int kb = kt * 32 + half * 8;
    const _Float16* hp = &hl16[wv][r][0];
    v16h a;
#pragma unroll
    for (int q = 0; q < 8; ++q) a[q] = hp[kb + q];
#pragma unroll
    for (int q = 0; q < 8; ++q) a[8 + q] = hp[kb + 16 + q];
    a2[kt] = a;
  }

  float partial[8];
#pragma unroll
  for (int j = 0; j < 8; ++j) partial[j] = 0.0f;

  // GEMM2: [16x64] @ [64x64]; scatter m, accumulate c partials
#pragma unroll
  for (int nt = 0; nt < 4; ++nt) {
    v8f c = {};
#pragma unroll
    for (int kt = 0; kt < 2; ++kt) {
      v16h b = load_b_frag_lds(ldsW2, kt, 64, nt * 16 + r, lane);
      c = __builtin_amdgcn_wmma_f32_16x16x32_f16(false, a2[kt], false, b, (short)0,
                                                 c, false, false);
    }
    int n = nt * 16 + r;
    float bb = b2[n];
    float wcn = wc[n];
#pragma unroll
    for (int j = 0; j < 8; ++j) {
      int m = j + half * 8;
      float v = silu_f(c[j] + bb);
      partial[j] += v * wcn;
      int recm = __shfl(rec, m, 32);
      bool vm = (base + m) < Minc;
      if (vm) atomicAdd(&msg_sum[(size_t)recm * CMSG + n], v);
    }
  }

  // reduce c over 64 columns: 4 per-lane + butterfly across 16-lane group
#pragma unroll
  for (int j = 0; j < 8; ++j) {
    float p = partial[j];
#pragma unroll
    for (int off = 1; off < 16; off <<= 1) p += __shfl_xor(p, off, 32);
    partial[j] = p + bc[0];
  }

#pragma unroll
  for (int j = 0; j < 8; ++j) {
    int m = j + half * 8;
    float dxm = __shfl(dxv, m, 32);
    float dym = __shfl(dyv, m, 32);
    float dzm = __shfl(dzv, m, 32);
    int recm = __shfl(rec, m, 32);
    bool vm = (base + m) < Minc;
    if (r == 0 && vm) {
      float cr = partial[j];
      atomicAdd(&dx_sum[(size_t)recm * 3 + 0], dxm * cr);
      atomicAdd(&dx_sum[(size_t)recm * 3 + 1], dym * cr);
      atomicAdd(&dx_sum[(size_t)recm * 3 + 2], dzm * cr);
      atomicAdd(&cnt[recm], 1.0f);
    }
  }
}

// ---------------------------------------------------------------------------
// x update: x_new = x + dxA/max(cntA,1) + dxB/max(cntB,1)
// ---------------------------------------------------------------------------
__global__ void x_update_kernel(const float* __restrict__ x,
                                const float* __restrict__ dxA, const float* __restrict__ cntA,
                                const float* __restrict__ dxB, const float* __restrict__ cntB,
                                float* __restrict__ xo, int n) {
  int i = blockIdx.x * blockDim.x + threadIdx.x;
  if (i >= n) return;
  float ca = cntA ? fmaxf(cntA[i], 1.0f) : 1.0f;
  float cb = cntB ? fmaxf(cntB[i], 1.0f) : 1.0f;
#pragma unroll
  for (int d = 0; d < 3; ++d) {
    float v = x[(size_t)i * 3 + d];
    if (dxA) v += dxA[(size_t)i * 3 + d] / ca;
    if (dxB) v += dxB[(size_t)i * 3 + d] / cb;
    xo[(size_t)i * 3 + d] = v;
  }
}

// ---------------------------------------------------------------------------
// Pooling kernels (monotone int encoding for float atomic max)
// ---------------------------------------------------------------------------
static __device__ __forceinline__ int enc_f32(float v) {
  int e = __float_as_int(v);
  return e >= 0 ? e : (e ^ 0x7fffffff);
}
static __device__ __forceinline__ float dec_f32(int e) {
  return __int_as_float(e >= 0 ? e : (e ^ 0x7fffffff));
}

__global__ void fill_i32_kernel(int* __restrict__ p, int v, int n) {
  int i = blockIdx.x * blockDim.x + threadIdx.x;
  if (i < n) p[i] = v;
}

__global__ void seg_max_kernel(const float* __restrict__ h, const int* __restrict__ batch,
                               int n, int* __restrict__ enc) {
  int idx = blockIdx.x * blockDim.x + threadIdx.x;
  if (idx >= n * CD_H) return;
  int i = idx >> 7, d = idx & 127;
  int b = batch[i];
  atomicMax(&enc[b * CD_H + d], enc_f32(h[idx]));
}

__global__ void seg_sum_kernel(const float* __restrict__ h, const int* __restrict__ batch,
                               int n, float* __restrict__ sum, float* __restrict__ cnt) {
  int idx = blockIdx.x * blockDim.x + threadIdx.x;
  if (idx >= n * CD_H) return;
  int i = idx >> 7, d = idx & 127;
  int b = batch[i];
  atomicAdd(&sum[b * CD_H + d], h[idx]);
  if (d == 0) atomicAdd(&cnt[b], 1.0f);
}

__global__ void build_H_kernel(const int* __restrict__ enc1, const int* __restrict__ enc2,
                               const float* __restrict__ sum3, const float* __restrict__ cnt3,
                               float* __restrict__ H) {
  int idx = blockIdx.x * blockDim.x + threadIdx.x;
  if (idx >= CB * 3 * CD_H) return;
  int b = idx / (3 * CD_H), d = idx % (3 * CD_H);
  float v;
  if (d < CD_H)            v = dec_f32(enc1[b * CD_H + d]);
  else if (d < 2 * CD_H)   v = dec_f32(enc2[b * CD_H + (d - CD_H)]);
  else                     v = sum3[b * CD_H + (d - 2 * CD_H)] / fmaxf(cnt3[b], 1.0f);
  H[idx] = v;
}

__global__ void log_softmax_kernel(const float* __restrict__ logits, float* __restrict__ out) {
  int b = blockIdx.x * blockDim.x + threadIdx.x;
  if (b >= CB) return;
  float mx = -3.4e38f;
#pragma unroll
  for (int k = 0; k < COUT; ++k) mx = fmaxf(mx, logits[b * COUT + k]);
  float s = 0.0f;
#pragma unroll
  for (int k = 0; k < COUT; ++k) s += __expf(logits[b * COUT + k] - mx);
  float lse = mx + __logf(s);
#pragma unroll
  for (int k = 0; k < COUT; ++k) out[b * COUT + k] = logits[b * COUT + k] - lse;
}

// ---------------------------------------------------------------------------
// Host orchestration
// ---------------------------------------------------------------------------
extern "C" void kernel_launch(void* const* d_in, const int* in_sizes, int n_in,
                              void* d_out, int out_size, void* d_ws, size_t ws_size,
                              hipStream_t stream) {
  (void)in_sizes; (void)n_in; (void)out_size; (void)ws_size;
  const int N1 = CN1, N2 = CN2, N3 = CN3, M1 = CM1, M2 = CM2;

  // ---- inputs (setup_inputs dict order; params jax-flattened: sorted keys) ----
  const float* h_in[3] = {(const float*)d_in[0], (const float*)d_in[1], (const float*)d_in[2]};
  const float* x_in[3] = {(const float*)d_in[3], (const float*)d_in[4], (const float*)d_in[5]};
  const int* b1 = (const int*)d_in[6];   // [2, M1] : row0 node idx, row1 edge idx
  const int* b2 = (const int*)d_in[7];   // [2, M2] : row0 edge idx, row1 tri idx
  const int* batch1 = (const int*)d_in[8];
  const int* batch2 = (const int*)d_in[9];
  const int* batch3 = (const int*)d_in[10];

  struct Lin { const float* w; const float* b; };
  int pi = 11;
  auto nextLin = [&]() { Lin l{(const float*)d_in[pi], (const float*)d_in[pi + 1]}; pi += 2; return l; };
  // params dict sorted: head < init < layers
  Lin head[3]; for (int k = 0; k < 3; ++k) head[k] = nextLin();
  Lin init[3]; for (int k = 0; k < 3; ++k) init[k] = nextLin();
  struct Lvl { Lin node1, node2, up_c, up_w1, up_w2, low_c, low_w1, low_w2; };
  Lvl L[4][3];
  for (int l = 0; l < 4; ++l) {
    // level 0 keys sorted: node1,node2,up_c,up_w1,up_w2
    L[l][0].node1 = nextLin(); L[l][0].node2 = nextLin();
    L[l][0].up_c = nextLin(); L[l][0].up_w1 = nextLin(); L[l][0].up_w2 = nextLin();
    // level 1: low_c,low_w1,low_w2,node1,node2,up_c,up_w1,up_w2
    L[l][1].low_c = nextLin(); L[l][1].low_w1 = nextLin(); L[l][1].low_w2 = nextLin();
    L[l][1].node1 = nextLin(); L[l][1].node2 = nextLin();
    L[l][1].up_c = nextLin(); L[l][1].up_w1 = nextLin(); L[l][1].up_w2 = nextLin();
    // level 2: low_c,low_w1,low_w2,node1,node2
    L[l][2].low_c = nextLin(); L[l][2].low_w1 = nextLin(); L[l][2].low_w2 = nextLin();
    L[l][2].node1 = nextLin(); L[l][2].node2 = nextLin();
  }

  // ---- workspace bump allocator ----
  char* ws = (char*)d_ws;
  size_t off = 0;
  auto alloc = [&](size_t bytes) -> void* {
    void* r = ws + off;
    off = (off + bytes + 255) & ~(size_t)255;
    return r;
  };

  // ---- weight packing (f16 B-fragment layout) ----
  auto packW = [&](const float* W, int K, int N) -> const _Float16* {
    int Np = (N + 15) & ~15;
    int KT = (K + 31) / 32;
    size_t total = (size_t)KT * Np * 32;
    _Float16* dst = (_Float16*)alloc(total * sizeof(_Float16));
    pack_b_kernel<<<(int)((total + 255) / 256), 256, 0, stream>>>(W, dst, K, N, Np, KT);
    return dst;
  };

  const _Float16* initp[3];
  for (int k = 0; k < 3; ++k) initp[k] = packW(init[k].w, CD_IN, CD_H);
  const _Float16* headp[3];
  headp[0] = packW(head[0].w, 3 * CD_H, 128);
  headp[1] = packW(head[1].w, 128, 256);
  headp[2] = packW(head[2].w, 256, COUT);

  struct LvlP { const _Float16 *up_w1p, *up_w2p, *low_w1p, *low_w2p, *node1p, *node2p; };
  LvlP P[4][3];
  for (int l = 0; l < 4; ++l) {
    for (int j = 0; j < 3; ++j) {
      int kcnt = (j < 2 ? 1 : 0) + (j > 0 ? 1 : 0);
      if (j < 2) {
        P[l][j].up_w1p = packW(L[l][j].up_w1.w, 2 * CD_H, CMSG);  // 256 rows; row 256 = rank-1
        P[l][j].up_w2p = packW(L[l][j].up_w2.w, CMSG, CMSG);
      }
      if (j > 0) {
        P[l][j].low_w1p = packW(L[l][j].low_w1.w, 2 * CD_H, CMSG);
        P[l][j].low_w2p = packW(L[l][j].low_w2.w, CMSG, CMSG);
      }
      P[l][j].node1p = packW(L[l][j].node1.w, CD_H + CMSG * kcnt, CMSG);
      P[l][j].node2p = packW(L[l][j].node2.w, CMSG, CD_H);
    }
  }

  // ---- activation / coordinate / message buffers ----
  const int NN[3] = {N1, N2, N3};
  float *hA[3], *hB[3], *xA[3], *xB[3];
  for (int k = 0; k < 3; ++k) {
    hA[k] = (float*)alloc((size_t)NN[k] * CD_H * 4);
    hB[k] = (float*)alloc((size_t)NN[k] * CD_H * 4);
    xA[k] = (float*)alloc((size_t)NN[k] * 3 * 4);
    xB[k] = (float*)alloc((size_t)NN[k] * 3 * 4);
  }
  float* msg_up0  = (float*)alloc((size_t)N1 * CMSG * 4);
  float* msg_up1  = (float*)alloc((size_t)N2 * CMSG * 4);
  float* msg_low1 = (float*)alloc((size_t)N2 * CMSG * 4);
  float* msg_low2 = (float*)alloc((size_t)N3 * CMSG * 4);
  float* dx_up0  = (float*)alloc((size_t)N1 * 3 * 4); float* cnt_up0  = (float*)alloc((size_t)N1 * 4);
  float* dx_up1  = (float*)alloc((size_t)N2 * 3 * 4); float* cnt_up1  = (float*)alloc((size_t)N2 * 4);
  float* dx_low1 = (float*)alloc((size_t)N2 * 3 * 4); float* cnt_low1 = (float*)alloc((size_t)N2 * 4);
  float* dx_low2 = (float*)alloc((size_t)N3 * 3 * 4); float* cnt_low2 = (float*)alloc((size_t)N3 * 4);
  float* tmp = (float*)alloc((size_t)N2 * CMSG * 4);
  int* enc1 = (int*)alloc((size_t)CB * CD_H * 4);
  int* enc2 = (int*)alloc((size_t)CB * CD_H * 4);
  float* sum3 = (float*)alloc((size_t)CB * CD_H * 4);
  float* cnt3 = (float*)alloc((size_t)CB * 4);
  float* Hbuf = (float*)alloc((size_t)CB * 3 * CD_H * 4);
  float* t0 = (float*)alloc((size_t)CB * 128 * 4);
  float* t1 = (float*)alloc((size_t)CB * 256 * 4);
  float* logitsB = (float*)alloc((size_t)CB * COUT * 4);

  // ---- launch helpers ----
  auto runGemm = [&](const float* S0, int w0, const float* S1, int w1,
                     const float* S2, int w2, const _Float16* Wp, const float* bias,
                     float* Y, int M, int N, int K, int act) {
    int Np = (N + 15) & ~15;
    int tm = (M + 15) / 16;
    if (Np == 64) {
      gemm_cat3_nt_kernel<4><<<(tm + 7) / 8, 256, 0, stream>>>(S0, w0, S1, w1, S2, w2,
                                                               Wp, bias, Y, M, N, K, act);
    } else if (Np == 128) {
      gemm_cat3_nt_kernel<8><<<(tm + 7) / 8, 256, 0, stream>>>(S0, w0, S1, w1, S2, w2,
                                                               Wp, bias, Y, M, N, K, act);
    } else if (Np == 16) {
      gemm_cat3_nt_kernel<1><<<(tm + 7) / 8, 256, 0, stream>>>(S0, w0, S1, w1, S2, w2,
                                                               Wp, bias, Y, M, N, K, act);
    } else {
      int tn = Np / 16;
      int blocks = (tm * tn + 7) / 8;
      gemm_cat3_gen_kernel<<<blocks, 256, 0, stream>>>(S0, w0, S1, w1, S2, w2, Wp, bias,
                                                       Y, M, N, Np, K, act);
    }
  };
  auto runMsg = [&](const float* hr, const float* hn, const float* xr, const float* xn,
                    const int* ridx, const int* nidx, int M,
                    const Lin& w1, const _Float16* w1p, const Lin& w2, const _Float16* w2p,
                    const Lin& cw, float* msum, float* dxs, float* cc) {
    int tiles = (M + 15) / 16;
    int blocks = (tiles + 7) / 8;
    message_kernel<<<blocks, 256, 0, stream>>>(hr, hn, xr, xn, ridx, nidx, M,
        w1p, w1.w + (size_t)(2 * CD_H) * CMSG, w1.b, w2p, w2.b, cw.w, cw.b,
        msum, dxs, cc);
  };

  // ---- init: h_k = h_in @ Winit + b ; x = x_in ----
  for (int k = 0; k < 3; ++k) {
    runGemm(h_in[k], CD_IN, nullptr, 0, nullptr, 0, initp[k], init[k].b,
            hA[k], NN[k], CD_H, CD_IN, 0);
    hipMemcpyAsync(xA[k], x_in[k], (size_t)NN[k] * 3 * 4, hipMemcpyDeviceToDevice, stream);
  }

  // ---- message passing layers ----
  for (int l = 0; l < 4; ++l) {
    hipMemsetAsync(msg_up0,  0, (size_t)N1 * CMSG * 4, stream);
    hipMemsetAsync(msg_up1,  0, (size_t)N2 * CMSG * 4, stream);
    hipMemsetAsync(msg_low1, 0, (size_t)N2 * CMSG * 4, stream);
    hipMemsetAsync(msg_low2, 0, (size_t)N3 * CMSG * 4, stream);
    hipMemsetAsync(dx_up0,  0, (size_t)N1 * 3 * 4, stream); hipMemsetAsync(cnt_up0,  0, (size_t)N1 * 4, stream);
    hipMemsetAsync(dx_up1,  0, (size_t)N2 * 3 * 4, stream); hipMemsetAsync(cnt_up1,  0, (size_t)N2 * 4, stream);
    hipMemsetAsync(dx_low1, 0, (size_t)N2 * 3 * 4, stream); hipMemsetAsync(cnt_low1, 0, (size_t)N2 * 4, stream);
    hipMemsetAsync(dx_low2, 0, (size_t)N3 * 3 * 4, stream); hipMemsetAsync(cnt_low2, 0, (size_t)N3 * 4, stream);

    // level 0 up: recv = b1 row0 (nodes), nbr = b1 row1 (edges)
    runMsg(hA[0], hA[1], xA[0], xA[1], b1, b1 + M1, M1,
           L[l][0].up_w1, P[l][0].up_w1p, L[l][0].up_w2, P[l][0].up_w2p, L[l][0].up_c,
           msg_up0, dx_up0, cnt_up0);
    // level 1 up: recv = b2 row0 (edges), nbr = b2 row1 (tris)
    runMsg(hA[1], hA[2], xA[1], xA[2], b2, b2 + M2, M2,
           L[l][1].up_w1, P[l][1].up_w1p, L[l][1].up_w2, P[l][1].up_w2p, L[l][1].up_c,
           msg_up1, dx_up1, cnt_up1);
    // level 1 low: recv = b1 row1 (edges), nbr = b1 row0 (nodes)
    runMsg(hA[1], hA[0], xA[1], xA[0], b1 + M1, b1, M1,
           L[l][1].low_w1, P[l][1].low_w1p, L[l][1].low_w2, P[l][1].low_w2p, L[l][1].low_c,
           msg_low1, dx_low1, cnt_low1);
    // level 2 low: recv = b2 row1 (tris), nbr = b2 row0 (edges)
    runMsg(hA[2], hA[1], xA[2], xA[1], b2 + M2, b2, M2,
           L[l][2].low_w1, P[l][2].low_w1p, L[l][2].low_w2, P[l][2].low_w2p, L[l][2].low_c,
           msg_low2, dx_low2, cnt_low2);

    // node updates: h_new = node2( silu(node1([h | msgs])) )
    runGemm(hA[0], CD_H, msg_up0, CMSG, nullptr, 0, P[l][0].node1p, L[l][0].node1.b,
            tmp, N1, CMSG, CD_H + CMSG, 1);
    runGemm(tmp, CMSG, nullptr, 0, nullptr, 0, P[l][0].node2p, L[l][0].node2.b,
            hB[0], N1, CD_H, CMSG, 0);
    runGemm(hA[1], CD_H, msg_up1, CMSG, msg_low1, CMSG, P[l][1].node1p, L[l][1].node1.b,
            tmp, N2, CMSG, CD_H + 2 * CMSG, 1);
    runGemm(tmp, CMSG, nullptr, 0, nullptr, 0, P[l][1].node2p, L[l][1].node2.b,
            hB[1], N2, CD_H, CMSG, 0);
    runGemm(hA[2], CD_H, msg_low2, CMSG, nullptr, 0, P[l][2].node1p, L[l][2].node1.b,
            tmp, N3, CMSG, CD_H + CMSG, 1);
    runGemm(tmp, CMSG, nullptr, 0, nullptr, 0, P[l][2].node2p, L[l][2].node2.b,
            hB[2], N3, CD_H, CMSG, 0);

    // x updates (each dx is segment-mean then summed)
    x_update_kernel<<<(N1 + 255) / 256, 256, 0, stream>>>(xA[0], dx_up0, cnt_up0,
                                                          nullptr, nullptr, xB[0], N1);
    x_update_kernel<<<(N2 + 255) / 256, 256, 0, stream>>>(xA[1], dx_up1, cnt_up1,
                                                          dx_low1, cnt_low1, xB[1], N2);
    x_update_kernel<<<(N3 + 255) / 256, 256, 0, stream>>>(xA[2], dx_low2, cnt_low2,
                                                          nullptr, nullptr, xB[2], N3);
    for (int k = 0; k < 3; ++k) {
      float* t = hA[k]; hA[k] = hB[k]; hB[k] = t;
      t = xA[k]; xA[k] = xB[k]; xB[k] = t;
    }
  }

  // ---- pooling: max(h1), max(h2), mean(h3) ----
  const int ENC_NEG = (int)0x807fffff;  // encode(-inf)
  fill_i32_kernel<<<(CB * CD_H + 255) / 256, 256, 0, stream>>>(enc1, ENC_NEG, CB * CD_H);
  fill_i32_kernel<<<(CB * CD_H + 255) / 256, 256, 0, stream>>>(enc2, ENC_NEG, CB * CD_H);
  hipMemsetAsync(sum3, 0, (size_t)CB * CD_H * 4, stream);
  hipMemsetAsync(cnt3, 0, (size_t)CB * 4, stream);
  seg_max_kernel<<<(int)(((size_t)N1 * CD_H + 255) / 256), 256, 0, stream>>>(hA[0], batch1, N1, enc1);
  seg_max_kernel<<<(int)(((size_t)N2 * CD_H + 255) / 256), 256, 0, stream>>>(hA[1], batch2, N2, enc2);
  seg_sum_kernel<<<(int)(((size_t)N3 * CD_H + 255) / 256), 256, 0, stream>>>(hA[2], batch3, N3, sum3, cnt3);
  build_H_kernel<<<(CB * 3 * CD_H + 255) / 256, 256, 0, stream>>>(enc1, enc2, sum3, cnt3, Hbuf);

  // ---- head MLP + log_softmax ----
  runGemm(Hbuf, 3 * CD_H, nullptr, 0, nullptr, 0, headp[0], head[0].b, t0, CB, 128, 3 * CD_H, 0);
  runGemm(t0, 128, nullptr, 0, nullptr, 0, headp[1], head[1].b, t1, CB, 256, 128, 2);
  runGemm(t1, 256, nullptr, 0, nullptr, 0, headp[2], head[2].b, logitsB, CB, COUT, 256, 0);
  log_softmax_kernel<<<1, CB, 0, stream>>>(logitsB, (float*)d_out);
}
